// HeadFusionAttention_54116587929966
// MI455X (gfx1250) — matrix operand order repaired and checked by reading, not compiled, not run on hardware
//
#include <hip/hip_runtime.h>

typedef __attribute__((ext_vector_type(16))) _Float16 v16h;
typedef __attribute__((ext_vector_type(8)))  _Float16 v8h;
typedef __attribute__((ext_vector_type(8)))  float    v8f;

#define BB   16
#define NN   1024
#define DIMD 1024
#define GG   4
#define GD   256
#define SCALE 0.08838834764831845f   // (1024/8)^-0.5

// ---------------------------------------------------------------------------
// WMMA helpers: D = A(16x32 f16) x B(32x16 f16) + C(16x16 f32)
// ---------------------------------------------------------------------------
__device__ __forceinline__ v8f wmma_f16(v16h a, v16h b, v8f c) {
  return __builtin_amdgcn_wmma_f32_16x16x32_f16(false, a, false, b,
                                                (short)0, c, false, false);
}

// 16-bit A-matrix 16x32 fragment (CDNA5 ISA 7.12.2):
// lane<16: row=lane,    K halves = {k0..k0+7, k0+16..k0+23}
// lane>=16: row=lane-16, K halves = {k0+8..k0+15, k0+24..k0+31}
__device__ __forceinline__ v16h load_a16x32(const _Float16* __restrict__ base,
                                            int ld, int row0, int k0) {
  int lane = threadIdx.x & 31;
  int r  = lane & 15;
  int ko = (lane < 16) ? 0 : 8;
  const _Float16* p = base + (size_t)(row0 + r) * ld + k0 + ko;
  union { v16h v; v8h h[2]; } u;
  u.h[0] = *(const v8h*)(p);
  u.h[1] = *(const v8h*)(p + 16);
  return u.v;
}

// 16-bit B-matrix 32x16 fragment: lane<16: col=lane, K={k0..k0+7, k0+16..k0+23};
// lane>=16: col=lane-16, K={k0+8..k0+15, k0+24..k0+31}.  src row-major [col][K].
__device__ __forceinline__ v16h load_b32x16(const _Float16* __restrict__ base,
                                            int ld, int col0, int k0) {
  int lane = threadIdx.x & 31;
  int c  = lane & 15;
  int ko = (lane < 16) ? 0 : 8;
  const _Float16* p = base + (size_t)(col0 + c) * ld + k0 + ko;
  union { v16h v; v8h h[2]; } u;
  u.h[0] = *(const v8h*)(p);
  u.h[1] = *(const v8h*)(p + 16);
  return u.v;
}

// ---------------------------------------------------------------------------
// Elementwise converts
// ---------------------------------------------------------------------------
__global__ void k_f32_to_f16(const float* __restrict__ src,
                             _Float16* __restrict__ dst, int n) {
  int i = blockIdx.x * blockDim.x + threadIdx.x;
  if (i < n) dst[i] = (_Float16)src[i];
}

// cur = x[:, :, grp*256 : (grp+1)*256] + head_x  (f16)
__global__ void k_make_cur(const float* __restrict__ x,
                           const float* __restrict__ hx,
                           _Float16* __restrict__ cur, int grp, int use_hx) {
  int i = blockIdx.x * blockDim.x + threadIdx.x;   // over B*N*GD
  int d = i & (GD - 1);
  int m = i >> 8;
  float v = x[(size_t)m * DIMD + grp * GD + d];
  if (use_hx) v += hx[i];
  cur[i] = (_Float16)v;
}

// ---------------------------------------------------------------------------
// QKV GEMM: (16384 x 256) x (768 x 256)^T.  One wave = 16x64 output
// (A-fragment reused across 4 N-tiles).  q pre-scaled; v stored transposed.
// ---------------------------------------------------------------------------
__global__ void __launch_bounds__(256)
k_qkv(const _Float16* __restrict__ cur, const _Float16* __restrict__ w,
      _Float16* __restrict__ q, _Float16* __restrict__ k,
      _Float16* __restrict__ vt) {
  int lane = threadIdx.x & 31;
  int wid = blockIdx.x * (blockDim.x >> 5) + (threadIdx.x >> 5);
  int mt = wid / 12;        // token tile (1024)
  int ng = wid % 12;        // group of 4 feature tiles (48/4)
  const v8f vz = {0.f, 0.f, 0.f, 0.f, 0.f, 0.f, 0.f, 0.f};
  v8f acc[4] = {vz, vz, vz, vz};
#pragma unroll
  for (int c = 0; c < 8; ++c) {
    v16h a = load_a16x32(cur, GD, mt * 16, c * 32);
#pragma unroll
    for (int u = 0; u < 4; ++u)
      acc[u] = wmma_f16(a, load_b32x16(w, GD, (ng * 4 + u) * 16, c * 32),
                        acc[u]);
  }
  int half8 = (lane < 16) ? 0 : 8;
  int mrow0 = mt * 16 + half8;
#pragma unroll
  for (int u = 0; u < 4; ++u) {
    int col = (ng * 4 + u) * 16 + (lane & 15);
    if (col < GD) {                                // Q (fold in softmax scale)
#pragma unroll
      for (int r = 0; r < 8; ++r)
        q[(size_t)(mrow0 + r) * GD + col] = (_Float16)(acc[u][r] * SCALE);
    } else if (col < 2 * GD) {                     // K
      int d = col - GD;
#pragma unroll
      for (int r = 0; r < 8; ++r)
        k[(size_t)(mrow0 + r) * GD + d] = (_Float16)acc[u][r];
    } else {                                       // V, transposed [b][d][n]
      int d  = col - 2 * GD;
      int b  = mt / 64;
      int n0 = (mt % 64) * 16 + half8;
      v8h pk;
#pragma unroll
      for (int r = 0; r < 8; ++r) pk[r] = (_Float16)acc[u][r];
      *(v8h*)(vt + ((size_t)b * GD + d) * NN + n0) = pk;
    }
  }
}

// ---------------------------------------------------------------------------
// Flash attention: one block (2 waves) per (batch, 16-query tile).
// Wave w covers keys [w*512, w*512+512); partials merged through LDS.
// Scores computed transposed (T = K_tile x Q^T) so lane = query m and the
// exp'd probabilities pack directly into the A-fragment layout for P x V.
// ---------------------------------------------------------------------------
__global__ void __launch_bounds__(64)
k_attn(const _Float16* __restrict__ qh, const _Float16* __restrict__ kh,
       const _Float16* __restrict__ vt, float* __restrict__ head_x,
       _Float16* __restrict__ concat, int grp) {
  __shared__ float lds_o[16 * 32 * 8];     // wave1 partial O (16KB)
  __shared__ float lds_max[32];
  __shared__ float lds_sum[32];

  int lane = threadIdx.x & 31;
  int wave = threadIdx.x >> 5;
  int wid = blockIdx.x;
  int b  = wid >> 6;
  int q0 = (wid & 63) * 16;
  const _Float16* Q = qh + (size_t)b * NN * GD;
  const _Float16* K = kh + (size_t)b * NN * GD;
  const _Float16* V = vt + (size_t)b * GD * NN;

  v16h qf[8];                          // Q as stationary B-fragments
#pragma unroll
  for (int c = 0; c < 8; ++c) qf[c] = load_b32x16(Q, GD, q0, c * 32);

  const v8f vz = {0.f, 0.f, 0.f, 0.f, 0.f, 0.f, 0.f, 0.f};
  v8f o[16];
#pragma unroll
  for (int t = 0; t < 16; ++t) o[t] = vz;

  float row_max = -1e30f, row_sum = 0.f;
  int src0 = (lane < 16) ? 0 : 8;
  int jbeg = wave * (NN / 2), jend = jbeg + (NN / 2);

  for (int j0 = jbeg; j0 < jend; j0 += 32) {
    if (j0 + 32 < jend)                 // prefetch next K slab
      __builtin_prefetch(K + (size_t)(j0 + 32 + lane) * GD, 0, 1);

    v8f t0 = vz, t1 = vz;              // T = K x Q^T; lane=m, vgpr=j
#pragma unroll
    for (int c = 0; c < 8; ++c) {
      v16h ka0 = load_a16x32(K, GD, j0,      c * 32);
      v16h ka1 = load_a16x32(K, GD, j0 + 16, c * 32);
      t0 = wmma_f16(ka0, qf[c], t0);
      t1 = wmma_f16(ka1, qf[c], t1);
    }
    // online softmax over this 32-key slab (row m = lane&15)
    float mx = row_max;
#pragma unroll
    for (int r = 0; r < 8; ++r) {
      mx = fmaxf(mx, t0[r]);
      mx = fmaxf(mx, t1[r]);
    }
    mx = fmaxf(mx, __shfl_xor(mx, 16, 32));
    float corr = __expf(row_max - mx);
    row_max = mx;

    v16h p;                            // P in A-fragment layout directly
    float s = 0.f;
#pragma unroll
    for (int r = 0; r < 8; ++r) {
      float e0 = __expf(t0[r] - mx);
      float e1 = __expf(t1[r] - mx);
      p[r]     = (_Float16)e0;
      p[r + 8] = (_Float16)e1;
      s += e0 + e1;
    }
    s += __shfl_xor(s, 16, 32);
    row_sum = row_sum * corr + s;

    float cf[8];                       // broadcast rescale factor per O-row
#pragma unroll
    for (int r = 0; r < 8; ++r) cf[r] = __shfl(corr, src0 + r, 32);
#pragma unroll
    for (int t = 0; t < 16; ++t)
#pragma unroll
      for (int r = 0; r < 8; ++r) o[t][r] *= cf[r];

#pragma unroll
    for (int t = 0; t < 16; ++t) {     // O += P x V over all 256 features
      v16h vb = load_b32x16(V, NN, t * 16, j0);
      o[t] = wmma_f16(p, vb, o[t]);
    }
  }

  // ---- merge the two waves' partials (flash combine) ----
  if (wave == 1) {
    lds_max[lane] = row_max;
    lds_sum[lane] = row_sum;
#pragma unroll
    for (int t = 0; t < 16; ++t)
      *(v8f*)&lds_o[(t * 32 + lane) * 8] = o[t];
  }
  __syncthreads();
  if (wave == 1) return;

  float m1 = lds_max[lane];
  float l1 = lds_sum[lane];
  float mx = fmaxf(row_max, m1);
  float c0 = __expf(row_max - mx);
  float c1 = __expf(m1 - mx);
  float L  = row_sum * c0 + l1 * c1;

  float cf0[8], cf1[8], inv[8];
#pragma unroll
  for (int r = 0; r < 8; ++r) {
    cf0[r] = __shfl(c0, src0 + r, 32);
    cf1[r] = __shfl(c1, src0 + r, 32);
    inv[r] = 1.f / __shfl(L, src0 + r, 32);
  }

  int dc = lane & 15;
  int m0 = q0 + src0;
#pragma unroll
  for (int t = 0; t < 16; ++t) {
    v8f o1 = *(const v8f*)&lds_o[(t * 32 + lane) * 8];
    int d = t * 16 + dc;
#pragma unroll
    for (int r = 0; r < 8; ++r) {
      float val = (o[t][r] * cf0[r] + o1[r] * cf1[r]) * inv[r];
      size_t m = (size_t)b * NN + m0 + r;
      head_x[m * GD + d] = val;                             // next group's add
      concat[m * DIMD + grp * GD + d] = (_Float16)val;      // proj input
    }
  }
}

// ---------------------------------------------------------------------------
// Projection: (16384 x 1024) x (1024 x 1024)^T + bias -> f32 out.
// One wave = 16x64 output (A reused across 4 N-tiles).
// ---------------------------------------------------------------------------
__global__ void __launch_bounds__(256)
k_proj(const _Float16* __restrict__ a, const _Float16* __restrict__ w,
       const float* __restrict__ bias, float* __restrict__ out) {
  int lane = threadIdx.x & 31;
  int wid = blockIdx.x * (blockDim.x >> 5) + (threadIdx.x >> 5);
  int mt = wid >> 4;        // token tile (1024)
  int ng = wid & 15;        // group of 4 feature tiles (64/4)
  const v8f vz = {0.f, 0.f, 0.f, 0.f, 0.f, 0.f, 0.f, 0.f};
  v8f acc[4] = {vz, vz, vz, vz};
  for (int k0 = 0; k0 < DIMD; k0 += 32) {
    v16h af = load_a16x32(a, DIMD, mt * 16, k0);
#pragma unroll
    for (int u = 0; u < 4; ++u)
      acc[u] = wmma_f16(af, load_b32x16(w, DIMD, (ng * 4 + u) * 16, k0),
                        acc[u]);
  }
  int mrow0 = mt * 16 + ((lane < 16) ? 0 : 8);
#pragma unroll
  for (int u = 0; u < 4; ++u) {
    int col = (ng * 4 + u) * 16 + (lane & 15);
    float bv = bias[col];
#pragma unroll
    for (int r = 0; r < 8; ++r)
      out[(size_t)(mrow0 + r) * DIMD + col] = acc[u][r] + bv;
  }
}

// ---------------------------------------------------------------------------
extern "C" void kernel_launch(void* const* d_in, const int* in_sizes, int n_in,
                              void* d_out, int out_size, void* d_ws,
                              size_t ws_size, hipStream_t stream) {
  (void)in_sizes; (void)n_in; (void)out_size; (void)ws_size;
  const float* x      = (const float*)d_in[0];
  const float* qkv_w  = (const float*)d_in[1];
  const float* proj_w = (const float*)d_in[2];
  const float* proj_b = (const float*)d_in[3];
  float* out = (float*)d_out;

  char* ws = (char*)d_ws;
  size_t off = 0;
  auto carve = [&](size_t bytes) -> char* {
    char* p = ws + off;
    off += (bytes + 255) & ~(size_t)255;
    return p;
  };
  float*    head_x = (float*)   carve((size_t)BB * NN * GD * 4);
  _Float16* cur_h  = (_Float16*)carve((size_t)BB * NN * GD * 2);
  _Float16* q_h    = (_Float16*)carve((size_t)BB * NN * GD * 2);
  _Float16* k_h    = (_Float16*)carve((size_t)BB * NN * GD * 2);
  _Float16* v_t    = (_Float16*)carve((size_t)BB * GD * NN * 2);
  _Float16* cat_h  = (_Float16*)carve((size_t)BB * NN * DIMD * 2);
  _Float16* wq_h   = (_Float16*)carve((size_t)GG * 3 * GD * GD * 2);
  _Float16* wp_h   = (_Float16*)carve((size_t)DIMD * DIMD * 2);

  const int nwq = GG * 3 * GD * GD;        // 786432
  const int nwp = DIMD * DIMD;             // 1048576
  k_f32_to_f16<<<(nwq + 255) / 256, 256, 0, stream>>>(qkv_w, wq_h, nwq);
  k_f32_to_f16<<<(nwp + 255) / 256, 256, 0, stream>>>(proj_w, wp_h, nwp);

  for (int g = 0; g < GG; ++g) {
    k_make_cur<<<(BB * NN * GD) / 256, 256, 0, stream>>>(x, head_x, cur_h, g,
                                                         g > 0 ? 1 : 0);
    // 1024 M-tiles * 12 N-groups waves, 8 waves/block
    k_qkv<<<1536, 256, 0, stream>>>(cur_h, wq_h + (size_t)g * 3 * GD * GD,
                                    q_h, k_h, v_t);
    // one 2-wave block per (batch, 16-query tile)
    k_attn<<<BB * (NN / 16), 64, 0, stream>>>(q_h, k_h, v_t, head_x, cat_h, g);
  }
  // 1024 M-tiles * 16 N-groups waves, 8 waves/block
  k_proj<<<2048, 256, 0, stream>>>(cat_h, wp_h, proj_b, out);
}
